// Decoder_51092930953358
// MI455X (gfx1250) — compile-verified, matrix operand order checked by reading, not run on hardware
//
#include <hip/hip_runtime.h>
#include <hip/hip_bf16.h>
#include <math.h>

typedef __attribute__((ext_vector_type(16))) _Float16 v16h;
typedef __attribute__((ext_vector_type(8)))  _Float16 v8h;
typedef __attribute__((ext_vector_type(8)))  float    v8f;

#define GF_BIAS   1
#define GF_RESID  2
#define GF_GELU   4
#define GF_TRANSB 8   // B is f32 stored (N,K) row-major (per-lane column contiguous)
#define GF_BH     16  // B is f16 stored (N,K) row-major (pre-transposed weights)

__device__ __forceinline__ v16h load_a16(const float* __restrict__ p)
{
    // p points at row start + k-base (incl. klo); elements e0..7 <- k+0..7,
    // e8..15 <- k+16..23 (CDNA5 f16 A/B per-lane layout)
    float4 a0 = *(const float4*)(p);
    float4 a1 = *(const float4*)(p + 4);
    float4 a2 = *(const float4*)(p + 16);
    float4 a3 = *(const float4*)(p + 20);
    v16h r;
    r[0]=(_Float16)a0.x; r[1]=(_Float16)a0.y; r[2]=(_Float16)a0.z; r[3]=(_Float16)a0.w;
    r[4]=(_Float16)a1.x; r[5]=(_Float16)a1.y; r[6]=(_Float16)a1.z; r[7]=(_Float16)a1.w;
    r[8]=(_Float16)a2.x; r[9]=(_Float16)a2.y; r[10]=(_Float16)a2.z; r[11]=(_Float16)a2.w;
    r[12]=(_Float16)a3.x; r[13]=(_Float16)a3.y; r[14]=(_Float16)a3.z; r[15]=(_Float16)a3.w;
    return r;
}

__device__ __forceinline__ v16h load_h16(const _Float16* __restrict__ p)
{
    v8h b0 = *(const v8h*)(p);
    v8h b1 = *(const v8h*)(p + 16);
    return __builtin_shufflevector(b0, b1, 0,1,2,3,4,5,6,7,8,9,10,11,12,13,14,15);
}

// ---------------------------------------------------------------------------
// Generic batched GEMM, C = A(MxK) * B(KxN) [+bias][+resid][gelu]
// A: fp32 -> f16 on load. MODE 0: B f32 (K,N); MODE 1: B f32 (N,K) (TRANSB);
// MODE 2: B f16 (N,K) pre-transposed. f32 accumulate, v_wmma_f32_16x16x32_f16,
// unroll-2 with dual accumulators to kill WMMA->WMMA RAW hazard NOPs.
// Batch z decomposed as (b = z/nh, h = z%nh). 128 thr = 4 waves, 16x16 C tile
// per wave.
// ---------------------------------------------------------------------------
template <int MODE>
__global__ __launch_bounds__(128)
void gemm_f32_wmma(const float* __restrict__ A, int lda, long sAb, long sAh,
                   const void* __restrict__ Bv, int ldb, long sBb, long sBh,
                   float* __restrict__ C, int ldc, long sCb, long sCh,
                   const float* __restrict__ bias,
                   const float* __restrict__ resid,
                   int M, int N, int K, int nh, int flags)
{
    const int wave = threadIdx.x >> 5;
    const int lane = threadIdx.x & 31;
    const int tn   = blockIdx.x * 4 + wave;
    if (tn * 16 >= N) return;
    const int tm = blockIdx.y;
    const int z  = blockIdx.z;
    const int zb = z / nh, zh = z - zb * nh;

    const float* Ab = A + (long)zb * sAb + (long)zh * sAh;
    float*       Cb = C + (long)zb * sCb + (long)zh * sCh;

    const int l15 = lane & 15;
    const int hi  = lane >> 4;            // 0 or 1
    const int klo = hi * 8;               // K sub-offset for this half-wave
    int am = tm * 16 + l15;               // A row this lane supplies
    if (am >= M) am = M - 1;              // clamp: dead D rows are never stored
    const int bn = tn * 16 + l15;         // B col / C col this lane supplies

    const float*    Arow  = Ab + (long)am * lda;
    const float*    Bf    = (const float*)Bv + (long)zb * sBb + (long)zh * sBh;
    const float*    BcolT = Bf + (long)bn * ldb;                  // MODE 1
    const _Float16* BrowH = (const _Float16*)Bv + (long)bn * ldb; // MODE 2 (ldb=K)

    auto loadB = [&](int kbase) -> v16h {
        if (MODE == 2) {
            return load_h16(BrowH + kbase);
        } else if (MODE == 1) {
            return load_a16(BcolT + kbase);
        } else {
            v16h bv;
            #pragma unroll
            for (int e = 0; e < 16; ++e) {
                int kk = kbase + (e < 8 ? e : e + 8);
                bv[e] = (_Float16)Bf[(long)kk * ldb + bn];   // coalesced over lanes
            }
            return bv;
        }
    };

    v8f acc0 = {}, acc1 = {};
    int k0 = 0;
    for (; k0 + 64 <= K; k0 += 64) {
        v16h a0 = load_a16(Arow + k0 + klo);
        v16h b0 = loadB(k0 + klo);
        v16h a1 = load_a16(Arow + k0 + 32 + klo);
        v16h b1 = loadB(k0 + 32 + klo);
        acc0 = __builtin_amdgcn_wmma_f32_16x16x32_f16(false, a0, false, b0, (short)0, acc0, false, false);
        acc1 = __builtin_amdgcn_wmma_f32_16x16x32_f16(false, a1, false, b1, (short)0, acc1, false, false);
    }
    if (k0 < K) {   // K % 64 == 32 tail (e.g. K = 1120)
        v16h a0 = load_a16(Arow + k0 + klo);
        v16h b0 = loadB(k0 + klo);
        acc0 = __builtin_amdgcn_wmma_f32_16x16x32_f16(false, a0, false, b0, (short)0, acc0, false, false);
    }

    const float bc = (flags & GF_BIAS) ? bias[bn] : 0.f;
    #pragma unroll
    for (int v = 0; v < 8; ++v) {
        int crow = tm * 16 + hi * 8 + v;
        if (crow < M) {
            float val = acc0[v] + acc1[v] + bc;
            if (flags & GF_RESID)
                val += resid[(long)zb * sCb + (long)zh * sCh + (long)crow * ldc + bn];
            if (flags & GF_GELU)
                val = 0.5f * val * (1.f + erff(val * 0.7071067811865475f));
            Cb[(long)crow * ldc + bn] = val;
        }
    }
}

// ---- weight transpose+convert: Bt[n*K+k] = (f16) Bsrc[k*N+n]
__global__ __launch_bounds__(256)
void wtrans_f16_k(const float* __restrict__ Bsrc, _Float16* __restrict__ Bt,
                  int K, int N, long tot)
{
    long i = blockIdx.x * 256L + threadIdx.x;
    if (i >= tot) return;
    int k = (int)(i % K); long n = i / K;
    Bt[i] = (_Float16)Bsrc[(long)k * N + n];
}

// ---- mean-pool over one axis
__global__ __launch_bounds__(256)
void meanpool_k(const float* __restrict__ in, float* __restrict__ out,
                long nout, long innerDiv, long outerStride, long redStride,
                int redCnt, float invCnt)
{
    long o = blockIdx.x * 256L + threadIdx.x;
    if (o >= nout) return;
    long outer = o / innerDiv, inner = o - outer * innerDiv;
    const float* p = in + outer * outerStride + inner;
    float s = 0.f;
    for (int k = 0; k < redCnt; ++k) s += p[(long)k * redStride];
    out[o] = s * invCnt;
}

// ---- patchify: Xp[b, h16*16+w16, (p1i*p2+p2i)*32+c] = pool[b, c, h16*p1+p1i, w16*p2+p2i]
__global__ __launch_bounds__(256)
void patchify_k(const float* __restrict__ pool, float* __restrict__ Xp,
                int R, int Cc, int p1, int p2, int F, long tot)
{
    long i = blockIdx.x * 256L + threadIdx.x;
    if (i >= tot) return;
    int f = (int)(i % F); long t = i / F;
    int n = (int)(t % 256); int b = (int)(t / 256);
    int c   = f % 32;
    int pp  = f / 32;
    int p2i = pp % p2, p1i = pp / p2;
    int rr = (n >> 4) * p1 + p1i;
    int cc = (n & 15) * p2 + p2i;
    Xp[i] = pool[(((long)(b * 32 + c) * R + rr) * Cc) + cc];
}

__global__ __launch_bounds__(256)
void addpos_k(float* __restrict__ t, const float* __restrict__ pos, long n, long mod)
{
    long i = blockIdx.x * 256L + threadIdx.x;
    if (i < n) t[i] += pos[i % mod];
}

// ---- LayerNorm over rows of 768, one wave per row
__global__ __launch_bounds__(256)
void layernorm_k(const float* __restrict__ x, float* __restrict__ y,
                 const float* __restrict__ g, const float* __restrict__ b, int rows)
{
    int row  = blockIdx.x * 8 + (threadIdx.x >> 5);
    int lane = threadIdx.x & 31;
    if (row >= rows) return;
    const float* xr = x + (long)row * 768;
    float s = 0.f, s2 = 0.f;
    for (int j = lane; j < 768; j += 32) { float v = xr[j]; s += v; s2 += v * v; }
    for (int off = 16; off > 0; off >>= 1) { s += __shfl_down(s, off); s2 += __shfl_down(s2, off); }
    s = __shfl(s, 0); s2 = __shfl(s2, 0);
    float mean = s * (1.f / 768.f);
    float var  = s2 * (1.f / 768.f) - mean * mean;
    float inv  = rsqrtf(var + 1e-5f);
    float* yr = y + (long)row * 768;
    for (int j = lane; j < 768; j += 32) yr[j] = (xr[j] - mean) * inv * g[j] + b[j];
}

// ---- in-place softmax over rows (scale applied pre-exp), one wave per row
__global__ __launch_bounds__(256)
void softmax_k(float* __restrict__ x, int rows, int cols, float scale)
{
    int row  = blockIdx.x * 8 + (threadIdx.x >> 5);
    int lane = threadIdx.x & 31;
    if (row >= rows) return;
    float* xr = x + (long)row * cols;
    float mx = -3.4e38f;
    for (int j = lane; j < cols; j += 32) mx = fmaxf(mx, xr[j] * scale);
    for (int off = 16; off > 0; off >>= 1) mx = fmaxf(mx, __shfl_down(mx, off));
    mx = __shfl(mx, 0);
    float s = 0.f;
    for (int j = lane; j < cols; j += 32) { float e = __expf(xr[j] * scale - mx); xr[j] = e; s += e; }
    for (int off = 16; off > 0; off >>= 1) s += __shfl_down(s, off);
    s = __shfl(s, 0);
    float r = 1.f / s;
    for (int j = lane; j < cols; j += 32) xr[j] *= r;
}

// ---- cross-attn degenerate einsum: s[z,j] = sum_i A[z,i,j]
__global__ __launch_bounds__(256)
void colsum_k(const float* __restrict__ S, float* __restrict__ s, int Z, int I, int J)
{
    long idx = blockIdx.x * 256L + threadIdx.x;
    if (idx >= (long)Z * J) return;
    int z = (int)(idx / J), j = (int)(idx % J);
    float acc = 0.f;
    const float* p = S + ((long)z * I) * J + j;
    for (int i = 0; i < I; ++i) acc += p[(long)i * J];
    s[idx] = acc;
}

// ---- o[b, j, h*64+d] = s[b,h,j] * vh[b, j, h*64+d]
__global__ __launch_bounds__(256)
void camul_k(const float* __restrict__ s, const float* __restrict__ vh,
             float* __restrict__ o, long n)
{
    long i = blockIdx.x * 256L + threadIdx.x;
    if (i >= n) return;
    int col = (int)(i % 768); long r = i / 768;
    int j = (int)(r % 256), b = (int)(r / 256);
    int h = col >> 6;
    o[i] = s[((long)(b * 12 + h)) * 256 + j] * vh[i];
}

// ---- fused unpatchify + broadcast combine + residual
__global__ __launch_bounds__(256)
void combine_k(const float* __restrict__ img, const float* __restrict__ Yax,
               const float* __restrict__ Ysag, const float* __restrict__ Ycor,
               const float* __restrict__ vw, float* __restrict__ out, long n)
{
    long i = blockIdx.x * 256L + threadIdx.x;
    if (i >= n) return;
    int w = (int)(i % 80); long t = i / 80;
    int h = (int)(t % 112); t /= 112;
    int d = (int)(t % 112); t /= 112;
    int c = (int)(t % 32);  int b = (int)(t / 32);
    float oax  = Yax [(long)b * 286720 + ((h / 7) * 16 + (w / 5)) * 1120 + ((h % 7) * 5 + (w % 5)) * 32 + c];
    float osag = Ysag[(long)b * 286720 + ((d / 7) * 16 + (w / 5)) * 1120 + ((d % 7) * 5 + (w % 5)) * 32 + c];
    float ocor = Ycor[(long)b * 401408 + ((d / 7) * 16 + (h / 7)) * 1568 + ((d % 7) * 7 + (h % 7)) * 32 + c];
    out[i] = img[i] + vw[0] * oax + vw[1] * osag + vw[2] * ocor;
}

// ---------------------------------------------------------------------------
extern "C" void kernel_launch(void* const* d_in, const int* in_sizes, int n_in,
                              void* d_out, int out_size, void* d_ws, size_t ws_size,
                              hipStream_t stream)
{
    (void)in_sizes; (void)n_in; (void)out_size; (void)ws_size;
    const float* img    = (const float*)d_in[0];
    const float* txt    = (const float*)d_in[1];
    const float* pe_w_ax  = (const float*)d_in[2];  const float* pe_b_ax  = (const float*)d_in[3];
    const float* pe_w_sag = (const float*)d_in[4];  const float* pe_b_sag = (const float*)d_in[5];
    const float* pe_w_cor = (const float*)d_in[6];  const float* pe_b_cor = (const float*)d_in[7];
    const float* pos_ax  = (const float*)d_in[8];
    const float* pos_sag = (const float*)d_in[9];
    const float* pos_cor = (const float*)d_in[10];
    const float* qw = (const float*)d_in[11];
    const float* kw = (const float*)d_in[12];
    const float* vw = (const float*)d_in[13];
    const float* pw = (const float*)d_in[14];  const float* pb = (const float*)d_in[15];
    const float* caq_w = (const float*)d_in[16]; const float* caq_b = (const float*)d_in[17];
    const float* cak_w = (const float*)d_in[18]; const float* cak_b = (const float*)d_in[19];
    const float* cav_w = (const float*)d_in[20]; const float* cav_b = (const float*)d_in[21];
    const float* cao_w = (const float*)d_in[22]; const float* cao_b = (const float*)d_in[23];
    const float* mlp_w1 = (const float*)d_in[24]; const float* mlp_b1 = (const float*)d_in[25];
    const float* mlp_w2 = (const float*)d_in[26]; const float* mlp_b2 = (const float*)d_in[27];
    const float* ow_ax  = (const float*)d_in[28]; const float* ob_ax  = (const float*)d_in[29];
    const float* ow_sag = (const float*)d_in[30]; const float* ob_sag = (const float*)d_in[31];
    const float* ow_cor = (const float*)d_in[32]; const float* ob_cor = (const float*)d_in[33];
    const float* view_w = (const float*)d_in[34];
    const float* n1_g = (const float*)d_in[35];
    const float* n2_g = (const float*)d_in[36];
    const float* n3_g = (const float*)d_in[37];
    const float* n1_b = (const float*)d_in[38];
    const float* n2_b = (const float*)d_in[39];
    const float* n3_b = (const float*)d_in[40];
    float* out = (float*)d_out;

    const float SC = 0.125f;             // 64^-0.5
    const long NTOK = 512;               // B*256 tokens

    // ---- workspace layout (float units, 16B-aligned chunks) ----
    float* wsp = (float*)d_ws;
    size_t off = 0;
    auto al  = [&](size_t nf) { float* p = wsp + off; off += (nf + 3) & ~(size_t)3; return p; };
    auto alh = [&](size_t nh_) { return (_Float16*)al((nh_ + 1) / 2); };
    float* pool_ax  = al(573440);   // (2,32,112,80)
    float* pool_sag = al(573440);
    float* pool_cor = al(802816);   // (2,32,112,112)
    float* Yax  = al(573440);       // (2,256,1120)
    float* Ysag = al(573440);
    float* Ycor = al(802816);       // (2,256,1568)
    float* qh   = al(154L * 768);   // cross-attn queries (branch-invariant)
    float* Xp = al(802816);
    float* t0 = al(393216); float* t1 = al(393216); float* t2 = al(393216);
    float* t3 = al(393216); float* t4 = al(393216); float* t5 = al(393216);
    float* t6 = al(393216);
    float* qb = al(393216); float* kb = al(393216); float* vb = al(393216);
    float* S    = al(1572864);      // (2,12,256,256)
    float* attnO = al(393216);
    float* khb = al(393216); float* vhb = al(393216);
    float* Sca = al(473088);        // (2,12,77,256)
    float* cs  = al(6144);          // (2,12,256)
    float* oca = al(393216);
    float* m1  = al(1572864);       // (512,3072)
    // f16 pre-transposed weights (N x K)
    _Float16* qwT  = alh(589824);  _Float16* kwT  = alh(589824);
    _Float16* vwT  = alh(589824);  _Float16* pwT  = alh(589824);
    _Float16* caqT = alh(589824);  _Float16* cakT = alh(589824);
    _Float16* cavT = alh(589824);  _Float16* caoT = alh(589824);
    _Float16* m1T  = alh(2359296); _Float16* m2T  = alh(2359296);
    _Float16* peT  = alh(1204224); // per-branch, reused (max 768 x 1568)
    _Float16* owT  = alh(1204224);

    auto ew = [&](long n) { return dim3((unsigned)((n + 255) / 256)); };
    auto gemm = [&](const float* A, int lda, long sAb, long sAh,
                    const void* B, int ldb, long sBb, long sBh,
                    float* C, int ldc, long sCb, long sCh,
                    const float* bias, const float* resid,
                    int M, int N, int K, int nbat, int nh, int flags) {
        dim3 grid((N + 63) / 64, (M + 15) / 16, nbat);
        if (flags & GF_BH)
            gemm_f32_wmma<2><<<grid, 128, 0, stream>>>(A, lda, sAb, sAh, B, ldb, sBb, sBh,
                                                       C, ldc, sCb, sCh, bias, resid, M, N, K, nh, flags);
        else if (flags & GF_TRANSB)
            gemm_f32_wmma<1><<<grid, 128, 0, stream>>>(A, lda, sAb, sAh, B, ldb, sBb, sBh,
                                                       C, ldc, sCb, sCh, bias, resid, M, N, K, nh, flags);
        else
            gemm_f32_wmma<0><<<grid, 128, 0, stream>>>(A, lda, sAb, sAh, B, ldb, sBb, sBh,
                                                       C, ldc, sCb, sCh, bias, resid, M, N, K, nh, flags);
    };
    auto wtrans = [&](const float* src, _Float16* dst, int K, int N) {
        long tot = (long)K * N;
        wtrans_f16_k<<<ew(tot), 256, 0, stream>>>(src, dst, K, N, tot);
    };

    // ---- pre-transpose branch-invariant weights to f16 (N x K) ----
    wtrans(qw, qwT, 768, 768);     wtrans(kw, kwT, 768, 768);
    wtrans(vw, vwT, 768, 768);     wtrans(pw, pwT, 768, 768);
    wtrans(caq_w, caqT, 768, 768); wtrans(cak_w, cakT, 768, 768);
    wtrans(cav_w, cavT, 768, 768); wtrans(cao_w, caoT, 768, 768);
    wtrans(mlp_w1, m1T, 768, 3072); wtrans(mlp_w2, m2T, 3072, 768);

    // ---- pooling (adaptive_avg_pool3d -> means over D / H / W) ----
    meanpool_k<<<ew(573440), 256, 0, stream>>>(img, pool_ax, 573440, 8960, 1003520, 8960, 112, 1.f / 112.f);
    meanpool_k<<<ew(573440), 256, 0, stream>>>(img, pool_sag, 573440, 80, 8960, 80, 112, 1.f / 112.f);
    meanpool_k<<<ew(802816), 256, 0, stream>>>(img, pool_cor, 802816, 1, 80, 1, 80, 1.f / 80.f);

    // ---- cross-attn queries from txt (branch-invariant) ----
    gemm(txt, 768, 0, 0, caqT, 768, 0, 0, qh, 768, 0, 0, caq_b, nullptr,
         154, 768, 768, 1, 1, GF_BIAS | GF_BH);

    auto branch = [&](const float* pool, int R, int Cc, int p1, int p2, int F,
                      const float* pe_w, const float* pe_b, const float* pos,
                      const float* ow, const float* ob, float* Y) {
        long totXp = NTOK * (long)F;
        wtrans(pe_w, peT, F, 768);   // (F,768) -> f16 (768,F)
        wtrans(ow, owT, 768, F);     // (768,F) -> f16 (F,768)
        patchify_k<<<ew(totXp), 256, 0, stream>>>(pool, Xp, R, Cc, p1, p2, F, totXp);
        // patch embed + pos
        gemm(Xp, F, 0, 0, peT, F, 0, 0, t0, 768, 0, 0, pe_b, nullptr,
             512, 768, F, 1, 1, GF_BIAS | GF_BH);
        addpos_k<<<ew(393216), 256, 0, stream>>>(t0, pos, 393216, 196608);
        layernorm_k<<<64, 256, 0, stream>>>(t0, t1, n1_g, n1_b, 512);
        // self-attention
        gemm(t1, 768, 0, 0, qwT, 768, 0, 0, qb, 768, 0, 0, nullptr, nullptr, 512, 768, 768, 1, 1, GF_BH);
        gemm(t1, 768, 0, 0, kwT, 768, 0, 0, kb, 768, 0, 0, nullptr, nullptr, 512, 768, 768, 1, 1, GF_BH);
        gemm(t1, 768, 0, 0, vwT, 768, 0, 0, vb, 768, 0, 0, nullptr, nullptr, 512, 768, 768, 1, 1, GF_BH);
        gemm(qb, 768, 196608, 64, kb, 768, 196608, 64, S, 256, 786432, 65536,
             nullptr, nullptr, 256, 256, 64, 24, 12, GF_TRANSB);                 // Q K^T
        softmax_k<<<(6144 + 7) / 8, 256, 0, stream>>>(S, 6144, 256, SC);
        gemm(S, 256, 786432, 65536, vb, 768, 196608, 64, attnO, 768, 196608, 64,
             nullptr, nullptr, 256, 64, 256, 24, 12, 0);                          // A V
        gemm(attnO, 768, 0, 0, pwT, 768, 0, 0, t2, 768, 0, 0, pb, t1,
             512, 768, 768, 1, 1, GF_BIAS | GF_RESID | GF_BH);                    // proj + residual
        layernorm_k<<<64, 256, 0, stream>>>(t2, t3, n2_g, n2_b, 512);
        // cross-attention (degenerate einsum)
        gemm(t3, 768, 0, 0, cakT, 768, 0, 0, khb, 768, 0, 0, cak_b, nullptr, 512, 768, 768, 1, 1, GF_BIAS | GF_BH);
        gemm(t3, 768, 0, 0, cavT, 768, 0, 0, vhb, 768, 0, 0, cav_b, nullptr, 512, 768, 768, 1, 1, GF_BIAS | GF_BH);
        gemm(qh, 768, 59136, 64, khb, 768, 196608, 64, Sca, 256, 236544, 19712,
             nullptr, nullptr, 77, 256, 64, 24, 12, GF_TRANSB);                   // Qtxt K^T
        softmax_k<<<(1848 + 7) / 8, 256, 0, stream>>>(Sca, 1848, 256, SC);
        colsum_k<<<ew(6144), 256, 0, stream>>>(Sca, cs, 24, 77, 256);
        camul_k<<<ew(393216), 256, 0, stream>>>(cs, vhb, oca, 393216);
        gemm(oca, 768, 0, 0, caoT, 768, 0, 0, t4, 768, 0, 0, cao_b, t3,
             512, 768, 768, 1, 1, GF_BIAS | GF_RESID | GF_BH);
        // MLP
        layernorm_k<<<64, 256, 0, stream>>>(t4, t5, n3_g, n3_b, 512);
        gemm(t5, 768, 0, 0, m1T, 768, 0, 0, m1, 3072, 0, 0, mlp_b1, nullptr,
             512, 3072, 768, 1, 1, GF_BIAS | GF_GELU | GF_BH);
        gemm(m1, 3072, 0, 0, m2T, 3072, 0, 0, t6, 768, 0, 0, mlp_b2, t4,
             512, 768, 3072, 1, 1, GF_BIAS | GF_RESID | GF_BH);
        // output projection
        gemm(t6, 768, 0, 0, owT, 768, 0, 0, Y, F, 0, 0, ob, nullptr,
             512, F, 768, 1, 1, GF_BIAS | GF_BH);
    };

    branch(pool_ax,  112,  80, 7, 5, 1120, pe_w_ax,  pe_b_ax,  pos_ax,  ow_ax,  ob_ax,  Yax);
    branch(pool_sag, 112,  80, 7, 5, 1120, pe_w_sag, pe_b_sag, pos_sag, ow_sag, ob_sag, Ysag);
    branch(pool_cor, 112, 112, 7, 7, 1568, pe_w_cor, pe_b_cor, pos_cor, ow_cor, ob_cor, Ycor);

    // ---- fused unpatchify + weighted broadcast + residual add ----
    const long ntot = 64225280;
    combine_k<<<ew(ntot), 256, 0, stream>>>(img, Yax, Ysag, Ycor, view_w, out, ntot);
}
